// DualDeformModule_42623255446049
// MI455X (gfx1250) — matrix operand order repaired and checked by reading, not compiled, not run on hardware
//
#include <hip/hip_runtime.h>

// ---------------------------------------------------------------------------
// DualDeformModule for MI455X (gfx1250, wave32, WMMA bf16 + TDM)
//   off_h   = relu(GN(conv3x3(reg_feat, w_oc)))
//   pts     = conv3x3(off_h, w_oo) + bias_oo
//   cls_out = relu(GN(deform_conv(cls_feat, pts, w_cls)))
//   reg_out = relu(GN(deform_conv(reg_feat, pts, w_reg)))
// (mask branch of the reference is dead code -> skipped)
//
// GEMM: M_tile=256 (all output channels), N_tile=64 pixels, K step 32.
// A (weights) pre-packed bf16 in fragment-major order; per step one 16KB
// contiguous TDM copy global->LDS overlapped with B-side gather/sampling.
// Each wave: 2 A-frags x 4 B-frags = 8 WMMAs per K step.
// ---------------------------------------------------------------------------

typedef __attribute__((ext_vector_type(16))) __bf16          v16bf;
typedef __attribute__((ext_vector_type(16))) unsigned short  v16u;
typedef __attribute__((ext_vector_type(8)))  unsigned short  v8u;
typedef __attribute__((ext_vector_type(8)))  float           v8f;
typedef __attribute__((ext_vector_type(4)))  unsigned int    u32x4;
typedef __attribute__((ext_vector_type(8)))  int             i32x8;
typedef __attribute__((ext_vector_type(4)))  int             i32x4;

#define BN    8
#define CC    256
#define HW    4096
#define NPIX  32768     // B*H*W
#define KTOT  2304      // 256*9 for both conv3x3 and deform einsum
#define KSTEPS 72       // KTOT / 32
#define ATILE 8192      // 256 rows x 32 k, bf16 elements (16KB)

__device__ __forceinline__ unsigned short f2bf(float f) {
  unsigned int u = __float_as_uint(f);
  u += 0x7FFFu + ((u >> 16) & 1u);          // round-to-nearest-even
  return (unsigned short)(u >> 16);
}
__device__ __forceinline__ float bf2f(unsigned short h) {
  return __uint_as_float(((unsigned int)h) << 16);
}

// ---------------------------------------------------------------------------
// TDM: contiguous copy of nelem bf16 global -> LDS, one issue per wave.
// D#: 2D tensor, data_size=2B, tile_dim0=nelem, tile_dim1=1.
// ---------------------------------------------------------------------------
__device__ __forceinline__ void tdm_copy(const unsigned short* gsrc,
                                         unsigned short* lds_dst, int nelem) {
  unsigned long long ga = (unsigned long long)(uintptr_t)gsrc;
  unsigned lds = (unsigned)(uintptr_t)lds_dst;   // low 32 bits = LDS offset
  u32x4 g0;
  g0[0] = 1u;                                        // count=1, no gather
  g0[1] = lds;                                       // lds_addr
  g0[2] = (unsigned)(ga & 0xFFFFFFFFu);              // global_addr[31:0]
  g0[3] = (unsigned)((ga >> 32) & 0x01FFFFFFu) | (2u << 30);  // [56:32]|type=2
  i32x8 g1;
  g1[0] = 1 << 16;                       // workgroup_mask=0, data_size=1 (2B)
  g1[1] = (nelem & 0xFFFF) << 16;        // tensor_dim0[15:0]   (bits 63:48)
  g1[2] = ((nelem >> 16) & 0xFFFF) | (1 << 16); // t_dim0[31:16] | t_dim1=1
  g1[3] = (nelem & 0xFFFF) << 16;        // tile_dim0           (bits 127:112)
  g1[4] = 1;                             // tile_dim1=1, tile_dim2=0
  g1[5] = nelem;                         // tensor_dim0_stride[31:0]
  g1[6] = 0;
  g1[7] = 0;
  i32x4 z4 = {};
#if defined(__clang_major__) && (__clang_major__ >= 23)
  i32x8 z8 = {};
  __builtin_amdgcn_tensor_load_to_lds(g0, g1, z4, z4, z8, 0);
#else
  __builtin_amdgcn_tensor_load_to_lds(g0, g1, z4, z4, 0);
#endif
}

// fp32 -> bf16 copy of a feature map
__global__ void cvt_bf16_kernel(const float* __restrict__ src,
                                unsigned short* __restrict__ dst, int n) {
  int e = blockIdx.x * blockDim.x + threadIdx.x;
  if (e < n) dst[e] = f2bf(src[e]);
}

// ---------------------------------------------------------------------------
// Pack weights -> bf16 fragment-major A tiles (one 16KB chunk per K step).
// dst[step*8192 + pos], pos -> (sub,slane,e):
//   row = sub*16 + (slane&15) ; k = (e&7) + 8*(slane>>4) + ((e&8)?16:0)
// MODE 0: kg = step*32+k (= c*9+r*3+s order)  -> wgt[row*KTOT + kg]
// MODE 1: kg = p*256+c GEMM order             -> wgt[row*KTOT + c*9 + p]
// ---------------------------------------------------------------------------
template<int MODE, int COUT>
__global__ void pack_weights_kernel(const float* __restrict__ wgt,
                                    unsigned short* __restrict__ dst,
                                    int total) {
  int idx = blockIdx.x * blockDim.x + threadIdx.x;
  if (idx >= total) return;
  int pos   = idx & (ATILE - 1);
  int step  = idx >> 13;
  int sub   = pos >> 9;                  // 0..15
  int slane = (pos >> 4) & 31;
  int e     = pos & 15;
  int row   = sub * 16 + (slane & 15);   // 0..255
  int k     = (e & 7) + 8 * (slane >> 4) + ((e & 8) ? 16 : 0);
  int kg    = step * 32 + k;
  float v = 0.f;
  if (row < COUT) {
    if (MODE == 0) {
      v = wgt[row * KTOT + kg];
    } else {
      int p = kg >> 8, c = kg & 255;
      v = wgt[row * KTOT + c * 9 + p];
    }
  }
  dst[idx] = f2bf(v);
}

// ---------------------------------------------------------------------------
// Implicit-GEMM WMMA kernel (one block = 256 channels x 64 pixels).
//   MODE 0 : conv3x3 im2col  (k = cin*9 + r*3 + s)
//   MODE 1 : deform sampling (k = p*256 + c), bilinear taps at (h+dy, w+dx)
// ---------------------------------------------------------------------------
template<int MODE, int COUT, bool OUT_BF16, bool HAS_BIAS, bool STATS>
__global__ __launch_bounds__(256)
void gemm_wmma_kernel(const unsigned short* __restrict__ xb,    // (8,256,64,64) bf16
                      const unsigned short* __restrict__ wpack, // packed A tiles
                      const float* __restrict__ aux,            // bias | pts
                      void* __restrict__ out,                   // (8,COUT,64,64)
                      float* __restrict__ gstats)               // (8,32,2)
{
  __shared__ __align__(32) unsigned short Alds[ATILE];
  __shared__ __align__(32) unsigned short Blds[2048];
  __shared__ float s_sum[32];
  __shared__ float s_sq[32];

  const int tid   = threadIdx.x;
  const int lane  = tid & 31;
  const int wave  = tid >> 5;
  const int Nbase = blockIdx.x * 64;     // 64 contiguous pixels = one image row
  const int bimg  = Nbase >> 12;
  const int half  = lane >> 4;
  const int lr    = lane & 15;

  // B-staging decomposition: 8 contiguous frag slots per thread
  const int pos      = tid * 8;          // 0..2040, 16B aligned
  const int bsub     = pos >> 9;
  const int slane    = (pos >> 4) & 31;
  const int ebase    = pos & 15;
  const int b_n      = bsub * 16 + (slane & 15);
  const int b_kstart = 8 * (slane >> 4) + 2 * ebase;
  const int b_ng     = Nbase + b_n;
  const int b_b      = b_ng >> 12;
  const int b_hw     = b_ng & 4095;
  const int b_h      = b_hw >> 6;
  const int b_w      = b_hw & 63;

  if (STATS && tid < 32) { s_sum[tid] = 0.f; s_sq[tid] = 0.f; }

  v8f acc[2][4];
  #pragma unroll
  for (int i = 0; i < 2; ++i)
    #pragma unroll
    for (int j = 0; j < 4; ++j) acc[i][j] = (v8f){};

  // persistent B-side index state (no divisions in the hot loop)
  int cin = 0, rs = 0;
  if (MODE == 0) { cin = b_kstart / 9; rs = b_kstart - 9 * cin; }
  float w00 = 0.f, w01 = 0.f, w10 = 0.f, w11 = 0.f;
  int   o00 = 0, o01 = 0, o10 = 0, o11 = 0;
  const unsigned short* imgc = xb;

  for (int k0 = 0; k0 < KTOT; k0 += 32) {
    __syncthreads();

    // ---- A tile: one 16KB TDM async copy (wave 0 only) --------------------
    if (wave == 0) tdm_copy(wpack + (size_t)(k0 >> 5) * ATILE, Alds, ATILE);

    // ---- stage B (32x64, fragment-major), overlapping the TDM -------------
    v8u buf;
    if (MODE == 0) {
      int cinL = cin, rsL = rs;
      #pragma unroll
      for (int i = 0; i < 8; ++i) {
        int r = rsL / 3, s = rsL - r * 3;        // rsL in [0,9)
        int y = b_h + r - 1, x = b_w + s - 1;
        bool inb = ((unsigned)y < 64u) & ((unsigned)x < 64u);
        int yc = min(max(y, 0), 63), xc = min(max(x, 0), 63);
        unsigned short raw = xb[(((b_b << 8) + cinL) << 12) + (yc << 6) + xc];
        buf[i] = inb ? raw : (unsigned short)0;
        rsL++; if (rsL == 9) { rsL = 0; cinL++; }
      }
      rs += 5; cin += 3; if (rs >= 9) { rs -= 9; cin++; }  // advance by 32
    } else {
      if ((k0 & 255) == 0) {               // new point p: refresh coords
        int p = k0 >> 8;
        float dy = aux[(((b_b * 18) + 2 * p)     << 12) + b_hw];
        float dx = aux[(((b_b * 18) + 2 * p + 1) << 12) + b_hw];
        float ys = (float)b_h + dy, xs = (float)b_w + dx;
        float y0f = floorf(ys), x0f = floorf(xs);
        float wy1 = ys - y0f, wx1 = xs - x0f;
        float wy0 = 1.f - wy1, wx0 = 1.f - wx1;
        int iy = (int)y0f, ix = (int)x0f;
        bool vy0 = (unsigned)iy       < 64u, vx0 = (unsigned)ix       < 64u;
        bool vy1 = (unsigned)(iy + 1) < 64u, vx1 = (unsigned)(ix + 1) < 64u;
        int y0c = min(max(iy,     0), 63), x0c = min(max(ix,     0), 63);
        int y1c = min(max(iy + 1, 0), 63), x1c = min(max(ix + 1, 0), 63);
        w00 = wy0 * wx0 * ((vy0 & vx0) ? 1.f : 0.f);
        w01 = wy0 * wx1 * ((vy0 & vx1) ? 1.f : 0.f);
        w10 = wy1 * wx0 * ((vy1 & vx0) ? 1.f : 0.f);
        w11 = wy1 * wx1 * ((vy1 & vx1) ? 1.f : 0.f);
        o00 = (y0c << 6) + x0c; o01 = (y0c << 6) + x1c;
        o10 = (y1c << 6) + x0c; o11 = (y1c << 6) + x1c;
        imgc = xb + ((size_t)((b_b << 8) + b_kstart) << 12);
      }
      const unsigned short* img = imgc;
      #pragma unroll
      for (int i = 0; i < 8; ++i) {        // 8 consecutive channels
        float a = bf2f(img[o00]) * w00 + bf2f(img[o01]) * w01 +
                  bf2f(img[o10]) * w10 + bf2f(img[o11]) * w11;
        buf[i] = f2bf(a);
        img += HW;
      }
      imgc += 32 * HW;                     // next step: +32 channels
    }
    *(v8u*)(Blds + pos) = buf;             // one ds_store_b128

    // ---- wait for TDM, then full barrier ----------------------------------
    if (wave == 0) __builtin_amdgcn_s_wait_tensorcnt(0);
    __syncthreads();

    // ---- 2 A-frags x 4 B-frags -> 8 WMMAs ---------------------------------
    v16u au0 = *(const v16u*)(Alds + (2 * wave)     * 512 + lane * 16);
    v16u au1 = *(const v16u*)(Alds + (2 * wave + 1) * 512 + lane * 16);
    v16bf a0 = __builtin_bit_cast(v16bf, au0);
    v16bf a1 = __builtin_bit_cast(v16bf, au1);
    #pragma unroll
    for (int j = 0; j < 4; ++j) {
      v16u bu = *(const v16u*)(Blds + j * 512 + lane * 16);
      v16bf b = __builtin_bit_cast(v16bf, bu);
      acc[0][j] = __builtin_amdgcn_wmma_f32_16x16x32_bf16(false, a0, false, b,
                                                (short)0, acc[0][j], false, false);
      acc[1][j] = __builtin_amdgcn_wmma_f32_16x16x32_bf16(false, a1, false, b,
                                                (short)0, acc[1][j], false, false);
    }
  }

  // ---- epilogue: bias, store, GN partial sums ----------------------------
  #pragma unroll
  for (int i = 0; i < 2; ++i) {
    float si = 0.f, sq = 0.f;
    #pragma unroll
    for (int r = 0; r < 8; ++r) {
      int ch = wave * 32 + i * 16 + r + 8 * half;  // C/D: M = vgpr + 8*half
      if (ch < COUT) {
        float bb = HAS_BIAS ? aux[ch] : 0.f;
        size_t base = ((size_t)(bimg * COUT + ch)) << 12;
        #pragma unroll
        for (int j = 0; j < 4; ++j) {
          float v = acc[i][j][r] + bb;
          int n = (Nbase + j * 16 + lr) & 4095;
          if (OUT_BF16) ((unsigned short*)out)[base + n] = f2bf(v);
          else          ((float*)out)[base + n] = v;
          if (STATS) { si += v; sq += v * v; }
        }
      }
    }
    if (STATS) {  // all r of one (wave,i,half) live in one group of 8 chans
      int g = wave * 4 + i * 2 + half;
      atomicAdd(&s_sum[g], si);
      atomicAdd(&s_sq[g],  sq);
    }
  }
  if (STATS) {
    __syncthreads();
    if (tid < 32) {
      atomicAdd(&gstats[((bimg << 5) + tid) * 2 + 0], s_sum[tid]);
      atomicAdd(&gstats[((bimg << 5) + tid) * 2 + 1], s_sq[tid]);
    }
  }
}

// ---------------------------------------------------------------------------
// In-place GroupNorm(affine) + ReLU.  n_per_group = 8 channels * 4096 = 32768
// ---------------------------------------------------------------------------
template<bool BF16IO>
__global__ void gn_apply_kernel(void* __restrict__ data,
                                const float* __restrict__ gstats,
                                const float* __restrict__ gamma,
                                const float* __restrict__ beta, int total) {
  int e = blockIdx.x * blockDim.x + threadIdx.x;
  if (e >= total) return;
  int ch = (e >> 12) & 255;
  int b  = e >> 20;
  int g  = ch >> 3;
  float s = gstats[((b << 5) + g) * 2 + 0];
  float q = gstats[((b << 5) + g) * 2 + 1];
  const float inv = 1.f / 32768.f;
  float m    = s * inv;
  float var  = q * inv - m * m;
  float rstd = rsqrtf(var + 1e-5f);
  float v = BF16IO ? bf2f(((unsigned short*)data)[e]) : ((float*)data)[e];
  v = (v - m) * rstd * gamma[ch] + beta[ch];
  v = fmaxf(v, 0.f);
  if (BF16IO) ((unsigned short*)data)[e] = f2bf(v);
  else        ((float*)data)[e] = v;
}

// ---------------------------------------------------------------------------
extern "C" void kernel_launch(void* const* d_in, const int* in_sizes, int n_in,
                              void* d_out, int out_size, void* d_ws, size_t ws_size,
                              hipStream_t stream) {
  (void)in_sizes; (void)n_in; (void)out_size; (void)ws_size;

  const float* cls_feat = (const float*)d_in[0];
  const float* reg_feat = (const float*)d_in[1];
  // d_in[2..6]: dead mask branch (w_mc, g_mc, b_mc, w_mo, bias_mo)
  const float* w_oc     = (const float*)d_in[7];
  const float* g_oc     = (const float*)d_in[8];
  const float* b_oc     = (const float*)d_in[9];
  const float* w_oo     = (const float*)d_in[10];
  const float* bias_oo  = (const float*)d_in[11];
  const float* w_cls    = (const float*)d_in[12];
  const float* g_cls    = (const float*)d_in[13];
  const float* bn_cls   = (const float*)d_in[14];
  const float* w_reg    = (const float*)d_in[15];
  const float* g_reg    = (const float*)d_in[16];
  const float* bn_reg   = (const float*)d_in[17];

  const int nfeat = BN * CC * HW;          // 8,388,608
  float* out_f = (float*)d_out;
  float* pts     = out_f;                  // (8,18,64,64)
  float* cls_out = out_f + 589824;         // (8,256,64,64)
  float* reg_out = out_f + 589824 + nfeat;

  // workspace carve-up
  unsigned short* cls_b = (unsigned short*)d_ws;
  unsigned short* reg_b = cls_b + nfeat;
  unsigned short* offh  = reg_b + nfeat;   // conv output, GN'd in place
  float* stats          = (float*)(offh + nfeat);
  float* st_off = stats;                   // 512 floats each
  float* st_cls = stats + 512;
  float* st_reg = stats + 1024;
  const int wsz = KSTEPS * ATILE;          // 589,824 bf16 per packed weight
  unsigned short* wA_oc  = (unsigned short*)(stats + 1536);
  unsigned short* wA_oo  = wA_oc + wsz;
  unsigned short* wA_cls = wA_oo + wsz;
  unsigned short* wA_reg = wA_cls + wsz;

  hipMemsetAsync(stats, 0, 1536 * sizeof(float), stream);

  dim3 blk(256);
  dim3 cvt_grid(nfeat / 256);
  cvt_bf16_kernel<<<cvt_grid, blk, 0, stream>>>(cls_feat, cls_b, nfeat);
  cvt_bf16_kernel<<<cvt_grid, blk, 0, stream>>>(reg_feat, reg_b, nfeat);

  // pack weights into fragment-major bf16 tiles (TDM source)
  dim3 pk_grid(wsz / 256);
  pack_weights_kernel<0, 256><<<pk_grid, blk, 0, stream>>>(w_oc, wA_oc, wsz);
  pack_weights_kernel<0, 18><<<pk_grid, blk, 0, stream>>>(w_oo, wA_oo, wsz);
  pack_weights_kernel<1, 256><<<pk_grid, blk, 0, stream>>>(w_cls, wA_cls, wsz);
  pack_weights_kernel<1, 256><<<pk_grid, blk, 0, stream>>>(w_reg, wA_reg, wsz);

  dim3 gemm_grid(NPIX / 64);

  // off_h raw conv (M=256) -> bf16 + GN stats
  gemm_wmma_kernel<0, 256, true, false, true>
      <<<gemm_grid, blk, 0, stream>>>(reg_b, wA_oc, nullptr, offh, st_off);
  // GN + ReLU in place on off_h
  gn_apply_kernel<true><<<cvt_grid, blk, 0, stream>>>(offh, st_off, g_oc, b_oc, nfeat);

  // pts = conv3x3(off_h, w_oo) + bias  (M=18, zero-padded to 256 rows)
  gemm_wmma_kernel<0, 18, false, true, false>
      <<<gemm_grid, blk, 0, stream>>>(offh, wA_oo, bias_oo, pts, nullptr);

  // deform convs (raw) + GN stats
  gemm_wmma_kernel<1, 256, false, false, true>
      <<<gemm_grid, blk, 0, stream>>>(cls_b, wA_cls, pts, cls_out, st_cls);
  gemm_wmma_kernel<1, 256, false, false, true>
      <<<gemm_grid, blk, 0, stream>>>(reg_b, wA_reg, pts, reg_out, st_reg);

  // GN + ReLU in place on outputs
  gn_apply_kernel<false><<<cvt_grid, blk, 0, stream>>>(cls_out, st_cls, g_cls, bn_cls, nfeat);
  gn_apply_kernel<false><<<cvt_grid, blk, 0, stream>>>(reg_out, st_reg, g_reg, bn_reg, nfeat);
}
